// VectorQuantizer_6708738916584
// MI455X (gfx1250) — matrix operand order repaired and checked by reading, not compiled, not run on hardware
//
#include <hip/hip_runtime.h>
#include <hip/hip_bf16.h>

// ---------------- problem constants (from reference) ----------------
#define N_ROWS   (32 * 64 * 64)   // 131072 z-vectors
#define CDIM     64               // embedding dim
#define NCODES   1024             // codebook entries
#define OUT_ELMS ((size_t)N_ROWS * CDIM)  // 8388608 floats of zq_st

typedef __attribute__((ext_vector_type(16))) __bf16 v16bf;
typedef __attribute__((ext_vector_type(8)))  float  v8f;

// fp32 -> bf16 with round-to-nearest-even
__device__ __forceinline__ __bf16 f2bf(float f) {
    unsigned u = __builtin_bit_cast(unsigned, f);
    u += 0x7FFFu + ((u >> 16) & 1u);
    unsigned short h = (unsigned short)(u >> 16);
    return __builtin_bit_cast(__bf16, h);
}
__device__ __forceinline__ float bf2f(__bf16 b) {
    unsigned short h = __builtin_bit_cast(unsigned short, b);
    unsigned u = ((unsigned)h) << 16;
    return __builtin_bit_cast(float, u);
}

// ------------------------------------------------------------------
// Prep: split codebook into bf16 hi/lo planes + bias b[n] = -0.5*||e_n||^2
// ------------------------------------------------------------------
__global__ void vq_prep(const float* __restrict__ embs,
                        __bf16* __restrict__ ehi,
                        __bf16* __restrict__ elo,
                        float*  __restrict__ bias) {
    int n = blockIdx.x * blockDim.x + threadIdx.x;
    if (n >= NCODES) return;
    const float* er = embs + (size_t)n * CDIM;
    float ss = 0.f;
#pragma unroll
    for (int k = 0; k < CDIM; ++k) {
        float x = er[k];
        ss += x * x;
        __bf16 h = f2bf(x);
        ehi[(size_t)n * CDIM + k] = h;
        elo[(size_t)n * CDIM + k] = f2bf(x - bf2f(h));
    }
    bias[n] = -0.5f * ss;
}

// ------------------------------------------------------------------
// Main: per wave, TWO 16-row A tiles (32 rows) x 1024 codes.
// Each B tile (16 codes, bf16 hi/lo) is loaded once and feeds 12
// v_wmma_f32_16x16x32_bf16 ops (bf16x3 split, K=64), doubling codebook
// reuse vs one tile per wave. Then argmax reduce, gather zq, loss.
// ------------------------------------------------------------------
__global__ __launch_bounds__(128) void vq_main(
    const float*  __restrict__ ze,
    const float*  __restrict__ embs,
    const __bf16* __restrict__ ehi,
    const __bf16* __restrict__ elo,
    const float*  __restrict__ bias,
    float*        __restrict__ out,
    float*        __restrict__ lossAcc)
{
    __shared__ int sIdx[4][32];

    const int lane = threadIdx.x & 31;
    const int wv   = threadIdx.x >> 5;
    const int g    = lane >> 4;      // half-wave group (0/1)
    const int m    = lane & 15;      // row (A) / column (B,C) within tile
    const int rowBase = (blockIdx.x * 4 + wv) * 32;

    // ---- Load 2 x 16 Z rows as bf16 hi/lo A fragments ----
    // A 16x32 layout (contiguous k, matches B): lanes 0-15: k = 0..15; lanes 16-31: k = 16..31
    v16bf ahi0[2], alo0[2], ahi1[2], alo1[2];
#pragma unroll
    for (int u = 0; u < 2; ++u) {
        const float*  zr = ze + (size_t)(rowBase + 16 * u + m) * CDIM;
        const float4* z4 = (const float4*)zr;
        float f0[16], f1[16];
        float4 a = z4[4 * g],      b = z4[4 * g + 1];   // k = 16g .. 16g+7
        float4 c = z4[4 * g + 2],  d = z4[4 * g + 3];   // k = 16g+8 .. 16g+15
        float4 e = z4[8 + 4 * g],  f = z4[9 + 4 * g];   // +32
        float4 p = z4[10 + 4 * g], q = z4[11 + 4 * g];  // +32
        f0[0]=a.x; f0[1]=a.y; f0[2]=a.z; f0[3]=a.w;
        f0[4]=b.x; f0[5]=b.y; f0[6]=b.z; f0[7]=b.w;
        f0[8]=c.x; f0[9]=c.y; f0[10]=c.z; f0[11]=c.w;
        f0[12]=d.x; f0[13]=d.y; f0[14]=d.z; f0[15]=d.w;
        f1[0]=e.x; f1[1]=e.y; f1[2]=e.z; f1[3]=e.w;
        f1[4]=f.x; f1[5]=f.y; f1[6]=f.z; f1[7]=f.w;
        f1[8]=p.x; f1[9]=p.y; f1[10]=p.z; f1[11]=p.w;
        f1[12]=q.x; f1[13]=q.y; f1[14]=q.z; f1[15]=q.w;
#pragma unroll
        for (int i = 0; i < 16; ++i) {
            __bf16 h0 = f2bf(f0[i]);
            ahi0[u][i] = h0; alo0[u][i] = f2bf(f0[i] - bf2f(h0));
            __bf16 h1 = f2bf(f1[i]);
            ahi1[u][i] = h1; alo1[u][i] = f2bf(f1[i] - bf2f(h1));
        }
    }

    float best[2][8];
    int   bidx[2][8];
#pragma unroll
    for (int u = 0; u < 2; ++u)
#pragma unroll
        for (int v = 0; v < 8; ++v) { best[u][v] = -3.0e38f; bidx[u][v] = 0; }

    // ---- Sweep all 64 code tiles; each B tile feeds both A tiles ----
    for (int t = 0; t < 64; ++t) {
        const int n = t * 16 + m;  // this lane's column (code index)
        const v16bf* bhp = (const v16bf*)(ehi + (size_t)n * CDIM + 16 * g);
        const v16bf* blp = (const v16bf*)(elo + (size_t)n * CDIM + 16 * g);
        v16bf bh0 = bhp[0], bh1 = bhp[2];
        v16bf bl0 = blp[0], bl1 = blp[2];
        float bv = bias[n];

#pragma unroll
        for (int u = 0; u < 2; ++u) {
            v8f c;
#pragma unroll
            for (int v = 0; v < 8; ++v) c[v] = bv;
            // bf16x3: hi*hi + hi*lo + lo*hi, K=64 => 2 WMMAs per term
            c = __builtin_amdgcn_wmma_f32_16x16x32_bf16(false, ahi0[u], false, bh0, (short)0, c, false, false);
            c = __builtin_amdgcn_wmma_f32_16x16x32_bf16(false, ahi1[u], false, bh1, (short)0, c, false, false);
            c = __builtin_amdgcn_wmma_f32_16x16x32_bf16(false, ahi0[u], false, bl0, (short)0, c, false, false);
            c = __builtin_amdgcn_wmma_f32_16x16x32_bf16(false, ahi1[u], false, bl1, (short)0, c, false, false);
            c = __builtin_amdgcn_wmma_f32_16x16x32_bf16(false, alo0[u], false, bh0, (short)0, c, false, false);
            c = __builtin_amdgcn_wmma_f32_16x16x32_bf16(false, alo1[u], false, bh1, (short)0, c, false, false);
#pragma unroll
            for (int v = 0; v < 8; ++v) {
                if (c[v] > best[u][v]) { best[u][v] = c[v]; bidx[u][v] = n; }
            }
        }
    }

    // ---- Argmax reduction across the 16 lanes of each column group ----
#pragma unroll
    for (int off = 1; off < 16; off <<= 1) {
#pragma unroll
        for (int u = 0; u < 2; ++u)
#pragma unroll
            for (int v = 0; v < 8; ++v) {
                float ov = __shfl_xor(best[u][v], off, 32);
                int   oi = __shfl_xor(bidx[u][v], off, 32);
                if (ov > best[u][v] || (ov == best[u][v] && oi < bidx[u][v])) {
                    best[u][v] = ov; bidx[u][v] = oi;
                }
            }
    }
    // lanes 0-15 hold rows v (v=0..7); lanes 16-31 hold rows v+8
    if (m == 0) {
#pragma unroll
        for (int u = 0; u < 2; ++u)
#pragma unroll
            for (int v = 0; v < 8; ++v) sIdx[wv][16 * u + 8 * g + v] = bidx[u][v];
    }
    __syncthreads();

    // ---- Gather zq + accumulate loss. Each lane: one full row (64 floats). ----
    {
        const int row  = rowBase + lane;
        const int code = sIdx[wv][lane];
        const float4* ev = (const float4*)(embs + (size_t)code * CDIM);
        const float4* zv = (const float4*)(ze   + (size_t)row  * CDIM);
        float4*       ov = (float4*)(out + (size_t)row * CDIM);
        float s = 0.f;
#pragma unroll
        for (int j = 0; j < 16; ++j) {
            float4 e = ev[j], z = zv[j];
            ov[j] = e;
            float dx = z.x - e.x, dy = z.y - e.y, dz = z.z - e.z, dw = z.w - e.w;
            s += dx * dx + dy * dy + dz * dz + dw * dw;
        }
#pragma unroll
        for (int off = 16; off; off >>= 1) s += __shfl_xor(s, off, 32);
        if (lane == 0)
            atomicAdd(lossAcc, s * (1.25f / (float)OUT_ELMS));
    }
}

// ------------------------------------------------------------------
extern "C" void kernel_launch(void* const* d_in, const int* in_sizes, int n_in,
                              void* d_out, int out_size, void* d_ws, size_t ws_size,
                              hipStream_t stream) {
    (void)in_sizes; (void)n_in; (void)out_size; (void)ws_size;
    const float* ze   = (const float*)d_in[0];
    const float* embs = (const float*)d_in[1];
    float* out = (float*)d_out;

    __bf16* ehi  = (__bf16*)d_ws;                    // 1024*64 bf16 = 128 KB
    __bf16* elo  = ehi + (size_t)NCODES * CDIM;      // 128 KB
    float*  bias = (float*)(elo + (size_t)NCODES * CDIM);  // 4 KB
    float*  lossAcc = out + OUT_ELMS;                // final loss scalar lives in d_out

    hipMemsetAsync(lossAcc, 0, sizeof(float), stream);
    vq_prep<<<NCODES / 128, 128, 0, stream>>>(embs, ehi, elo, bias);
    vq_main<<<N_ROWS / 128, 128, 0, stream>>>(ze, embs, ehi, elo, bias, out, lossAcc);
}
